// AdaGeo_47871705481379
// MI455X (gfx1250) — matrix operand order repaired and checked by reading, not compiled, not run on hardware
//
#include <hip/hip_runtime.h>
#include <hip/hip_bf16.h>
#include <math.h>

typedef __attribute__((ext_vector_type(16))) _Float16 v16h;
typedef __attribute__((ext_vector_type(8)))  _Float16 v8h;
typedef __attribute__((ext_vector_type(8)))  float    v8f;

#define N1V 4096
#define N2V 2048
#define PDV 256
#define PD2V 258
#define FDV 772
#define KP1 288     // padded K for 258-dim GEMMs (mult of 32)
#define NP1 272     // padded N for 258-dim operands (mult of 16)
#define FPAD 800    // padded K for 772-dim GEMM (mult of 32)
#define EPSV 1e-12f

union V16U { v16h v; v8h h[2]; };

// ---------------- block reductions ----------------
__device__ __forceinline__ float block_red_sum(float v, float* buf) {
  int t = threadIdx.x;
  buf[t] = v; __syncthreads();
  for (int s = blockDim.x >> 1; s > 0; s >>= 1) {
    if (t < s) buf[t] += buf[t + s];
    __syncthreads();
  }
  float r = buf[0]; __syncthreads();
  return r;
}
__device__ __forceinline__ float block_red_max(float v, float* buf) {
  int t = threadIdx.x;
  buf[t] = v; __syncthreads();
  for (int s = blockDim.x >> 1; s > 0; s >>= 1) {
    if (t < s) buf[t] = fmaxf(buf[t], buf[t + s]);
    __syncthreads();
  }
  float r = buf[0]; __syncthreads();
  return r;
}

// ---------------- generic WMMA GEMM: C = (A[M,K] * B[N,K]^T + bias) * oscale ----------------
// A,B f16 row-major (K contiguous). One of Cf (f32) / Ch (f16) is non-null.
// Output row stride ldc, column offset coff, columns written only if n < Nout.
// M mult of 16; K mult of 32; lda/ldb mult of 8 (16B-aligned rows).
// NB = N-tiles per wave (register blocking; A fragment reused NB times).
// K-loop is software-pipelined: fragments for k+32 are in flight while WMMAs run on k.
template<int NB>
__global__ __launch_bounds__(256) void wmma_gemm_k(
    const _Float16* __restrict__ A, int lda,
    const _Float16* __restrict__ B, int ldb,
    const float* __restrict__ bias,
    float* __restrict__ Cf, _Float16* __restrict__ Ch,
    int ldc, int coff, int Nout, int K, float oscale, int ntx)
{
  const int wave = threadIdx.x >> 5;
  const int lane = threadIdx.x & 31;
  const int g0   = (blockIdx.x * 8 + wave) * NB;   // first 16-wide tile owned by this wave
  if (g0 >= ntx) return;                           // wave-uniform
  const int mbase = blockIdx.y << 4;
  const int l15   = lane & 15;
  const int hl    = lane >> 4;

  const _Float16* arow = A + (size_t)(mbase + l15) * lda;
  const _Float16* brow[NB];
#pragma unroll
  for (int t = 0; t < NB; ++t) {
    int tile = g0 + t;
    if (tile > ntx - 1) tile = ntx - 1;            // clamp tail (store masked later)
    brow[t] = B + (size_t)((tile << 4) + l15) * ldb;
  }

  v8f acc[NB];
#pragma unroll
  for (int t = 0; t < NB; ++t) acc[t] = (v8f){};

  // ---- prologue: load fragments for k0 = 0 ----
  // A frag (16x32 MxK): lane holds row l15; K = hl*16 + e (16 contiguous halves)
  // B frag (32x16 KxN): lane holds col l15; K = hl*16 + e (16 contiguous halves)
  V16U a_cur, b_cur[NB];
  a_cur.h[0] = *(const v8h*)(arow + hl * 16);
  a_cur.h[1] = *(const v8h*)(arow + hl * 16 + 8);
#pragma unroll
  for (int t = 0; t < NB; ++t) {
    b_cur[t].h[0] = *(const v8h*)(brow[t] + hl * 16);
    b_cur[t].h[1] = *(const v8h*)(brow[t] + hl * 16 + 8);
  }

  // ---- steady state: prefetch k0, compute k0-32 ----
  for (int k0 = 32; k0 < K; k0 += 32) {
    V16U a_nxt, b_nxt[NB];
    a_nxt.h[0] = *(const v8h*)(arow + k0 + hl * 16);
    a_nxt.h[1] = *(const v8h*)(arow + k0 + hl * 16 + 8);
#pragma unroll
    for (int t = 0; t < NB; ++t) {
      b_nxt[t].h[0] = *(const v8h*)(brow[t] + k0 + hl * 16);
      b_nxt[t].h[1] = *(const v8h*)(brow[t] + k0 + hl * 16 + 8);
    }
#pragma unroll
    for (int t = 0; t < NB; ++t)
      acc[t] = __builtin_amdgcn_wmma_f32_16x16x32_f16(
          false, a_cur.v, false, b_cur[t].v, (short)0, acc[t], false, false);
    a_cur = a_nxt;
#pragma unroll
    for (int t = 0; t < NB; ++t) b_cur[t] = b_nxt[t];
  }

  // ---- epilogue: last k-step ----
#pragma unroll
  for (int t = 0; t < NB; ++t)
    acc[t] = __builtin_amdgcn_wmma_f32_16x16x32_f16(
        false, a_cur.v, false, b_cur[t].v, (short)0, acc[t], false, false);

#pragma unroll
  for (int t = 0; t < NB; ++t) {
    const int n = ((g0 + t) << 4) + l15;
    if (g0 + t < ntx && n < Nout) {
      const float bz = bias ? bias[n] : 0.0f;
#pragma unroll
      for (int r = 0; r < 8; ++r) {
        const int m = mbase + (hl << 3) + r;       // C layout: VGPR r -> row r + 8*hl
        const float v = (acc[t][r] + bz) * oscale;
        const size_t idx = (size_t)m * ldc + coff + n;
        if (Cf) Cf[idx] = v;
        else    Ch[idx] = (_Float16)v;
      }
    }
  }
}

// ---------------- conversions ----------------
__global__ void cvt_f16_k(_Float16* __restrict__ dst, const float* __restrict__ src, int n) {
  int i = blockIdx.x * blockDim.x + threadIdx.x;
  if (i < n) dst[i] = (_Float16)src[i];
}

__global__ void pad_cvt_k(_Float16* __restrict__ dst, const float* __restrict__ src,
                          int R, int C, int Rs, int Cs) {
  int i = blockIdx.x * blockDim.x + threadIdx.x;
  if (i >= R * C) return;
  int r = i / C, c = i % C;
  float v = (r < Rs && c < Cs) ? src[(size_t)r * Cs + c] : 0.0f;
  dst[i] = (_Float16)v;
}

// lmfT[c][i] = lm_feature[i][c], padded to 272 rows, f16
__global__ void lmft_k(_Float16* __restrict__ dst,
                       const float* __restrict__ lmX, const float* __restrict__ lmY) {
  int idx = blockIdx.x * blockDim.x + threadIdx.x;   // over 272*4096
  if (idx >= NP1 * N1V) return;
  int c = idx >> 12, i = idx & (N1V - 1);
  float v = 0.0f;
  if (c < PDV) v = lmX[(size_t)i * PDV + c];
  else if (c < PD2V) v = lmY[(size_t)i * 2 + (c - PDV)];
  dst[idx] = (_Float16)v;
}

// ---------------- router / delay kernels ----------------
// block b = feature column c; reduce over 4096 landmarks
__global__ __launch_bounds__(256) void router_reduce_k(
    float* __restrict__ router0, float* __restrict__ dsdot, float* __restrict__ dssum,
    const float* __restrict__ lmX, const float* __restrict__ lmY,
    const float* __restrict__ lm_delay,
    const float* __restrict__ g1, const float* __restrict__ alpha, const float* __restrict__ beta)
{
  __shared__ float buf[256];
  int c = blockIdx.x;
  float a = alpha[0], b0 = beta[0], g = g1[0];
  float sf = 0.f, sdf = 0.f, sd = 0.f;
  for (int i = threadIdx.x; i < N1V; i += 256) {
    float f = (c < PDV) ? lmX[(size_t)i * PDV + c] : lmY[(size_t)i * 2 + (c - PDV)];
    float ds = expf(-g * (a * lm_delay[i] + b0));
    sf += f; sdf += ds * f; sd += ds;
  }
  float tf  = block_red_sum(sf, buf);
  float tdf = block_red_sum(sdf, buf);
  float td  = block_red_sum(sd, buf);
  if (threadIdx.x == 0) {
    router0[c] = tf / (float)N1V;
    dsdot[c]   = tdf;
    if (c == 0) dssum[0] = td;
  }
}

__global__ void rp0_k(float* __restrict__ rp0, const float* __restrict__ router0,
                      const float* __restrict__ dsdot, const float* __restrict__ dssum) {
  int c = threadIdx.x;
  if (c < PD2V) rp0[c] = (dsdot[c] + router0[c]) / (1.0f + dssum[0] + EPSV);
}

__global__ void rou_k(float* __restrict__ rou0, float* __restrict__ rou1,
                      const float* __restrict__ tg_delay,
                      const float* __restrict__ g2, const float* __restrict__ g3,
                      const float* __restrict__ alpha, const float* __restrict__ beta) {
  int i = blockIdx.x * blockDim.x + threadIdx.x;
  if (i >= N2V) return;
  float t = alpha[0] * tg_delay[i] + beta[0];
  rou0[i] = expf(-g2[0] * t);
  rou1[i] = expf(-g3[0] * t);
}

// router_1 = router_prop0 @ w1_w^T + w1_b  (tiny 258x258, f32 VALU)
__global__ void router1_k(float* __restrict__ r1, const float* __restrict__ rp0,
                          const float* __restrict__ w1w, const float* __restrict__ w1b) {
  int c = threadIdx.x;
  if (c >= PD2V) return;
  float s = w1b[c];
  for (int k = 0; k < PD2V; ++k) s += rp0[k] * w1w[(size_t)c * PD2V + k];
  r1[c] = s;
}

// ---------------- softmax / attr (stores attr-1 = expm1(softmax) in f16) ----------------
__global__ __launch_bounds__(256) void attr_k(
    _Float16* __restrict__ attr, float* __restrict__ deg,
    const _Float16* __restrict__ scores, const float* __restrict__ rou0)
{
  __shared__ float buf[256];
  int m = blockIdx.x;
  const _Float16* row = scores + (size_t)m * N1V;
  float mx = -1e30f;
  for (int i = threadIdx.x; i < N1V; i += 256) mx = fmaxf(mx, (float)row[i]);
  mx = block_red_max(mx, buf);
  float se = 0.f;
  for (int i = threadIdx.x; i < N1V; i += 256) se += expf((float)row[i] - mx);
  se = block_red_sum(se, buf);
  float rinv = 1.0f / se;
  float sa = 0.f;
  for (int i = threadIdx.x; i < N1V; i += 256) {
    float p  = expf((float)row[i] - mx) * rinv;  // softmax
    float am = expm1f(p);                        // attr - 1
    attr[(size_t)m * N1V + i] = (_Float16)am;
    sa += am;
  }
  sa = block_red_sum(sa, buf);
  if (threadIdx.x == 0)
    deg[m] = 1.0f + (sa + (float)N1V) + rou0[m] + EPSV;  // 1 + sum(attr) + rou0 + eps
}

// tg_prop0 (f16, padded 288): ((attr-1)@lmf + 4096*router0 + tgf0 + rou0*router0)/deg
__global__ void prop0_k(_Float16* __restrict__ p0h, const float* __restrict__ raw,
                        const float* __restrict__ tgX, const float* __restrict__ rou0,
                        const float* __restrict__ router0, const float* __restrict__ deg) {
  int idx = blockIdx.x * blockDim.x + threadIdx.x;   // 2048*288
  if (idx >= N2V * KP1) return;
  int m = idx / KP1, c = idx % KP1;
  float v = 0.0f;
  if (c < PD2V) {
    float t0 = (c < PDV) ? tgX[(size_t)m * PDV + c] : 0.0f;
    float num = raw[(size_t)m * NP1 + c] + ((float)N1V + rou0[m]) * router0[c] + t0;
    v = num / deg[m];
  }
  p0h[idx] = (_Float16)v;
}

// tg_prop1 (f16, padded 288): (tg_f1 + rou1*router_1)/(1+rou1+eps)
__global__ void prop1_k(_Float16* __restrict__ p1h, const float* __restrict__ outf,
                        const float* __restrict__ rou1, const float* __restrict__ r1) {
  int idx = blockIdx.x * blockDim.x + threadIdx.x;   // 2048*288
  if (idx >= N2V * KP1) return;
  int m = idx / KP1, c = idx % KP1;
  float v = 0.0f;
  if (c < PD2V) {
    float f1 = outf[(size_t)m * FDV + PDV + c];
    v = (f1 + rou1[m] * r1[c]) / (1.0f + rou1[m] + EPSV);
  }
  p1h[idx] = (_Float16)v;
}

// write tg_X into d_out final cols 0..255; gather f16 padded copy of final feature (K=800)
__global__ void assemble_k(_Float16* __restrict__ finalh, float* __restrict__ outf,
                           const float* __restrict__ tgX) {
  int idx = blockIdx.x * blockDim.x + threadIdx.x;   // 2048*800
  if (idx >= N2V * FPAD) return;
  int m = idx / FPAD, c = idx % FPAD;
  float v = 0.0f;
  if (c < PDV)       { v = tgX[(size_t)m * PDV + c]; outf[(size_t)m * FDV + c] = v; }
  else if (c < FDV)  { v = outf[(size_t)m * FDV + c]; }
  finalh[idx] = (_Float16)v;
}

// fused softmax2 + y_pred (never materializes attn2)
__global__ __launch_bounds__(256) void ypred_k(
    float* __restrict__ y, const _Float16* __restrict__ scores2,
    const float* __restrict__ lmY, const float* __restrict__ pvw, const float* __restrict__ pvb)
{
  __shared__ float buf[256];
  int m = blockIdx.x;
  const _Float16* row = scores2 + (size_t)m * N1V;
  float mx = -1e30f;
  for (int i = threadIdx.x; i < N1V; i += 256) mx = fmaxf(mx, (float)row[i]);
  mx = block_red_max(mx, buf);
  float w00 = pvw[0], w01 = pvw[1], w10 = pvw[2], w11 = pvw[3];
  float b0 = pvb[0], b1 = pvb[1];
  float se = 0.f, s0 = 0.f, s1 = 0.f;
  for (int i = threadIdx.x; i < N1V; i += 256) {
    float e  = expf((float)row[i] - mx);
    float y0 = lmY[(size_t)i * 2], y1 = lmY[(size_t)i * 2 + 1];
    se += e;
    s0 += e * (y0 * w00 + y1 * w01 + b0);
    s1 += e * (y0 * w10 + y1 * w11 + b1);
  }
  se = block_red_sum(se, buf);
  s0 = block_red_sum(s0, buf);
  s1 = block_red_sum(s1, buf);
  if (threadIdx.x == 0) {
    y[(size_t)m * 2]     = s0 / se;
    y[(size_t)m * 2 + 1] = s1 / se;
  }
}

// ---------------- host launcher ----------------
extern "C" void kernel_launch(void* const* d_in, const int* in_sizes, int n_in,
                              void* d_out, int out_size, void* d_ws, size_t ws_size,
                              hipStream_t stream) {
  (void)in_sizes; (void)n_in; (void)out_size; (void)ws_size;
  const float* lm_X  = (const float*)d_in[0];
  const float* lm_Y  = (const float*)d_in[1];
  const float* tg_X  = (const float*)d_in[2];
  const float* lm_dl = (const float*)d_in[4];
  const float* tg_dl = (const float*)d_in[5];
  const float* aq_w = (const float*)d_in[6];  const float* aq_b = (const float*)d_in[7];
  const float* ak_w = (const float*)d_in[8];  const float* ak_b = (const float*)d_in[9];
  const float* w1_w = (const float*)d_in[10]; const float* w1_b = (const float*)d_in[11];
  const float* w2_w = (const float*)d_in[12]; const float* w2_b = (const float*)d_in[13];
  const float* pq_w = (const float*)d_in[14]; const float* pq_b = (const float*)d_in[15];
  const float* pk_w = (const float*)d_in[16]; const float* pk_b = (const float*)d_in[17];
  const float* pv_w = (const float*)d_in[18]; const float* pv_b = (const float*)d_in[19];
  const float* g1 = (const float*)d_in[20];
  const float* g2 = (const float*)d_in[21];
  const float* g3 = (const float*)d_in[22];
  const float* al = (const float*)d_in[23];
  const float* be = (const float*)d_in[24];

  float* y_out     = (float*)d_out;
  float* out_final = y_out + (size_t)N2V * 2;

  // workspace sub-allocation (256B aligned)
  char* wsp = (char*)d_ws; size_t off = 0;
  auto alloc = [&](size_t bytes) -> void* {
    void* p = wsp + off; off = (off + bytes + 255) & ~(size_t)255; return p;
  };
  _Float16* lmX_h  = (_Float16*)alloc((size_t)N1V * PDV * 2);
  _Float16* tgX_h  = (_Float16*)alloc((size_t)N2V * PDV * 2);
  _Float16* aqw_h  = (_Float16*)alloc((size_t)128 * PDV * 2);
  _Float16* akw_h  = (_Float16*)alloc((size_t)128 * PDV * 2);
  _Float16* pkw_h  = (_Float16*)alloc((size_t)128 * PDV * 2);
  _Float16* w1_h   = (_Float16*)alloc((size_t)NP1 * KP1 * 2);
  _Float16* w2_h   = (_Float16*)alloc((size_t)NP1 * KP1 * 2);
  _Float16* pqw_h  = (_Float16*)alloc((size_t)128 * FPAD * 2);
  _Float16* lmfT_h = (_Float16*)alloc((size_t)NP1 * N1V * 2);
  _Float16* q_h    = (_Float16*)alloc((size_t)N2V * 128 * 2);
  _Float16* k_h    = (_Float16*)alloc((size_t)N1V * 128 * 2);
  _Float16* q2_h   = (_Float16*)alloc((size_t)N2V * 128 * 2);
  _Float16* k2_h   = (_Float16*)alloc((size_t)N1V * 128 * 2);
  _Float16* sc_h   = (_Float16*)alloc((size_t)N2V * N1V * 2);   // scores, reused for scores2
  _Float16* attr_h = (_Float16*)alloc((size_t)N2V * N1V * 2);   // attr - 1
  float*    p0raw  = (float*)alloc((size_t)N2V * NP1 * 4);
  _Float16* p0_h   = (_Float16*)alloc((size_t)N2V * KP1 * 2);
  _Float16* p1_h   = (_Float16*)alloc((size_t)N2V * KP1 * 2);
  _Float16* fin_h  = (_Float16*)alloc((size_t)N2V * FPAD * 2);
  float* router0 = (float*)alloc(NP1 * 4);
  float* dsdot   = (float*)alloc(NP1 * 4);
  float* dssum   = (float*)alloc(256);
  float* rp0     = (float*)alloc(NP1 * 4);
  float* r1      = (float*)alloc(NP1 * 4);
  float* rou0    = (float*)alloc(N2V * 4);
  float* rou1    = (float*)alloc(N2V * 4);
  float* deg     = (float*)alloc(N2V * 4);

  const float invT = 1.0f / sqrtf(128.0f);

  // nb: 1 = one 16-wide tile per wave (small N), 4 = 64-wide per wave (big N)
  auto gemm = [&](int nb, const _Float16* A, int lda, const _Float16* B, int ldb,
                  const float* bias, float* Cf, _Float16* Ch, int ldc, int coff,
                  int Nout, int M, int K, float oscale, int ntx) {
    if (nb == 4) {
      dim3 g((ntx + 31) / 32, M / 16);
      wmma_gemm_k<4><<<g, dim3(256), 0, stream>>>(A, lda, B, ldb, bias, Cf, Ch,
                                                  ldc, coff, Nout, K, oscale, ntx);
    } else {
      dim3 g((ntx + 7) / 8, M / 16);
      wmma_gemm_k<1><<<g, dim3(256), 0, stream>>>(A, lda, B, ldb, bias, Cf, Ch,
                                                  ldc, coff, Nout, K, oscale, ntx);
    }
  };
  auto cvt = [&](_Float16* dst, const float* src, int n) {
    cvt_f16_k<<<(n + 255) / 256, 256, 0, stream>>>(dst, src, n);
  };

  // ---- conversions ----
  cvt(lmX_h, lm_X, N1V * PDV);
  cvt(tgX_h, tg_X, N2V * PDV);
  cvt(aqw_h, aq_w, 128 * PDV);
  cvt(akw_h, ak_w, 128 * PDV);
  cvt(pkw_h, pk_w, 128 * PDV);
  pad_cvt_k<<<(NP1 * KP1 + 255) / 256, 256, 0, stream>>>(w1_h, w1_w, NP1, KP1, PD2V, PD2V);
  pad_cvt_k<<<(NP1 * KP1 + 255) / 256, 256, 0, stream>>>(w2_h, w2_w, NP1, KP1, PD2V, PD2V);
  pad_cvt_k<<<(128 * FPAD + 255) / 256, 256, 0, stream>>>(pqw_h, pq_w, 128, FPAD, 128, FDV);
  lmft_k<<<(NP1 * N1V + 255) / 256, 256, 0, stream>>>(lmfT_h, lm_X, lm_Y);

  // ---- router / delay scalars ----
  router_reduce_k<<<PD2V, 256, 0, stream>>>(router0, dsdot, dssum, lm_X, lm_Y, lm_dl, g1, al, be);
  rp0_k<<<1, 288, 0, stream>>>(rp0, router0, dsdot, dssum);
  rou_k<<<(N2V + 255) / 256, 256, 0, stream>>>(rou0, rou1, tg_dl, g2, g3, al, be);

  // ---- attention 1: q,k proj; scores; attr ----
  gemm(1, tgX_h, PDV, aqw_h, PDV, aq_b, nullptr, q_h, 128, 0, 128, N2V, PDV, invT, 8);
  gemm(1, lmX_h, PDV, akw_h, PDV, ak_b, nullptr, k_h, 128, 0, 128, N1V, PDV, 1.0f, 8);
  gemm(4, q_h, 128, k_h, 128, nullptr, nullptr, sc_h, N1V, 0, N1V, N2V, 128, 1.0f, N1V / 16);
  attr_k<<<N2V, 256, 0, stream>>>(attr_h, deg, sc_h, rou0);

  // ---- propagation 0: (attr-1) @ lm_feature^T, then fixup ----
  gemm(4, attr_h, N1V, lmfT_h, N1V, nullptr, p0raw, nullptr, NP1, 0, NP1, N2V, N1V, 1.0f, NP1 / 16);
  prop0_k<<<(N2V * KP1 + 255) / 256, 256, 0, stream>>>(p0_h, p0raw, tg_X, rou0, router0, deg);
  router1_k<<<1, 288, 0, stream>>>(r1, rp0, w1_w, w1_b);

  // ---- tg_feature_1 -> d_out cols [256,514) ----
  gemm(4, p0_h, KP1, w1_h, KP1, w1_b, out_final, nullptr, FDV, PDV, PD2V, N2V, KP1, 1.0f, NP1 / 16);
  prop1_k<<<(N2V * KP1 + 255) / 256, 256, 0, stream>>>(p1_h, out_final, rou1, r1);
  // ---- tg_feature_2 -> d_out cols [514,772) ----
  gemm(4, p1_h, KP1, w2_h, KP1, w2_b, out_final, nullptr, FDV, PDV + PD2V, PD2V, N2V, KP1, 1.0f, NP1 / 16);

  // ---- assemble final feature (also writes tg_X part of d_out) ----
  assemble_k<<<(N2V * FPAD + 255) / 256, 256, 0, stream>>>(fin_h, out_final, tg_X);

  // ---- attention 2: q2,k2 proj; scores2; fused softmax+y ----
  gemm(1, fin_h, FPAD, pqw_h, FPAD, pq_b, nullptr, q2_h, 128, 0, 128, N2V, FPAD, invT, 8);
  gemm(1, lmX_h, PDV, pkw_h, PDV, pk_b, nullptr, k2_h, 128, 0, 128, N1V, PDV, 1.0f, 8);
  gemm(4, q2_h, 128, k2_h, 128, nullptr, nullptr, sc_h, N1V, 0, N1V, N2V, 128, 1.0f, N1V / 16);
  ypred_k<<<N2V, 256, 0, stream>>>(y_out, sc_h, lm_Y, pv_w, pv_b);
}
